// WindowAttention2D_29678224015611
// MI455X (gfx1250) — matrix operand order repaired and checked by reading, not compiled, not run on hardware
//
#include <hip/hip_runtime.h>

typedef __attribute__((ext_vector_type(16))) _Float16 v16h;
typedef __attribute__((ext_vector_type(8)))  float    v8f;

#define WIN_H 32
#define SHIFT 16
#define NH    8
#define HD    64
#define CDIM  512
#define TDIM  1248
#define DDIM  128
#define NWH   4

// 16-bit WMMA A-fragment K index for vector slot j, lane group g (ISA 7.12.2)
__device__ __forceinline__ int amapK(int j, int g) {
  return (j < 8) ? (j + 8 * g) : (j + 8 + 8 * g);
}

// Pre-swizzle an f32 [512 x N] weight matrix into f16 WMMA-B fragment order:
// dst flat index = ((t*16 + kt)*32 + lane)*16 + j ; element = W[kt*32 + amapK(j, g)][t*16 + (lane&15)]
__global__ void swizzle_w_f16(const float* __restrict__ W, int N, _Float16* __restrict__ dst) {
  int o    = blockIdx.x * 256 + threadIdx.x;
  int j    = o & 15;
  int lane = (o >> 4) & 31;
  int kt   = (o >> 9) & 15;
  int t    = o >> 13;
  int g    = lane >> 4;
  int row  = kt * 32 + amapK(j, g);
  int col  = t * 16 + (lane & 15);
  dst[o] = (_Float16)W[(size_t)row * N + col];
}

__global__ void __launch_bounds__(256)
win_attn_fused(const float* __restrict__ x,
               const _Float16* __restrict__ wQKV,   // [96][16][32][16] swizzled f16
               const _Float16* __restrict__ wProj,  // [32][16][32][16] swizzled f16
               const float* __restrict__ bProj,
               const float* __restrict__ relTab,    // [19593][8]
               const float* __restrict__ qnW,
               const float* __restrict__ knW,
               float* __restrict__ out)
{
  __shared__ __align__(32) _Float16 sX[2 * 16 * 32 * 16]; // A-swizzled X; reused for attn-out
  __shared__ __align__(32) _Float16 sQ[32 * CDIM];
  __shared__ __align__(32) _Float16 sK[32 * CDIM];
  __shared__ __align__(32) _Float16 sV[32 * CDIM];
  __shared__ __align__(32) _Float16 sP[NH * 32 * 32];
  __shared__ float sQs[32 * NH];
  __shared__ float sKs[32 * NH];

  const int tid  = threadIdx.x;
  const int lane = tid & 31;
  const int wave = tid >> 5;
  const int lm   = lane & 15;
  const int g    = lane >> 4;
  const int rcol = blockIdx.x;   // 0..1247
  const int wH   = blockIdx.y;   // 0..3

  // ---- Phase 0: gather rolled window rows, f32->f16, store in WMMA-A swizzle
  for (int e = 0; e < 16; ++e) {
    int idx4 = e * 256 + tid;            // 4096 float4s of the 32x512 tile
    int m    = idx4 >> 7;                // token
    int c0   = (idx4 & 127) << 2;        // channel base
    int d    = (wH * WIN_H + m + SHIFT) & (DDIM - 1);
    const float4 v = *reinterpret_cast<const float4*>(
        &x[(((size_t)d * TDIM) + rcol) * CDIM + c0]);
    float vals[4] = {v.x, v.y, v.z, v.w};
#pragma unroll
    for (int q = 0; q < 4; ++q) {
      int c  = c0 + q;
      int kk = c & 31, ks = c >> 5;
      int ga = (kk >> 3) & 1;
      int j  = (kk & 7) + ((kk & 16) >> 1);
      sX[(((m >> 4) * 16 + ks) * 32 + ((m & 15) + 16 * ga)) * 16 + j] = (_Float16)vals[q];
    }
  }
  __syncthreads();

  // ---- Phase 1: QKV = X @ Wqkv   (M=32, N=1536, K=512)
  for (int t = wave; t < 96; t += 8) {
    v8f acc0 = {}, acc1 = {};
    const _Float16* bb = wQKV + (size_t)t * (16 * 32 * 16);
#pragma unroll 4
    for (int kt = 0; kt < 16; ++kt) {
      v16h a0 = *reinterpret_cast<const v16h*>(&sX[((kt) * 32 + lane) * 16]);
      v16h a1 = *reinterpret_cast<const v16h*>(&sX[((16 + kt) * 32 + lane) * 16]);
      v16h b  = *reinterpret_cast<const v16h*>(&bb[(kt * 32 + lane) * 16]);
      acc0 = __builtin_amdgcn_wmma_f32_16x16x32_f16(false, a0, false, b, (short)0, acc0, false, false);
      acc1 = __builtin_amdgcn_wmma_f32_16x16x32_f16(false, a1, false, b, (short)0, acc1, false, false);
    }
    int n0 = t * 16;
    _Float16* dst; int col;
    if (n0 < 512)       { dst = sQ; col = n0; }
    else if (n0 < 1024) { dst = sK; col = n0 - 512; }
    else                { dst = sV; col = n0 - 1024; }
#pragma unroll
    for (int r = 0; r < 8; ++r) {
      int m = r + 8 * g;
      dst[(size_t)(m)      * CDIM + col + lm] = (_Float16)acc0[r];
      dst[(size_t)(m + 16) * CDIM + col + lm] = (_Float16)acc1[r];
    }
  }
  __syncthreads();

  // ---- Phase 2: RMSNorm scales per (token, head)
  {
    int tok = tid >> 3, h = tid & 7;
    int base = tok * CDIM + h * HD;
    float sq = 0.f, sk = 0.f;
#pragma unroll 8
    for (int d = 0; d < HD; ++d) {
      float qv = (float)sQ[base + d];
      float kv = (float)sK[base + d];
      sq += qv * qv; sk += kv * kv;
    }
    sQs[tok * NH + h] = rsqrtf(sq * (1.f / HD) + 1e-6f);
    sKs[tok * NH + h] = rsqrtf(sk * (1.f / HD) + 1e-6f);
  }
  __syncthreads();

  // ---- Phase 3: attention; wave == head
  {
    const int h = wave;
    v16h qa[2][2], kb[2][2];
#pragma unroll
    for (int mt = 0; mt < 2; ++mt)
#pragma unroll
      for (int ks = 0; ks < 2; ++ks) {
        int m = mt * 16 + lm;
        float sc = sQs[m * NH + h] * 0.125f;   // fold hd^-0.5
#pragma unroll
        for (int j = 0; j < 16; ++j) {
          int d = ks * 32 + amapK(j, g);
          qa[mt][ks][j] = (_Float16)((float)sQ[m * CDIM + h * HD + d] * sc * qnW[d]);
        }
      }
#pragma unroll
    for (int nt = 0; nt < 2; ++nt)
#pragma unroll
      for (int ks = 0; ks < 2; ++ks) {
        int n = nt * 16 + lm;
        float sc = sKs[n * NH + h];
#pragma unroll
        for (int j = 0; j < 16; ++j) {
          int d = ks * 32 + amapK(j, g);       // B-fragment K map
          kb[nt][ks][j] = (_Float16)((float)sK[n * CDIM + h * HD + d] * sc * knW[d]);
        }
      }

    v8f S[2][2];
#pragma unroll
    for (int mt = 0; mt < 2; ++mt)
#pragma unroll
      for (int nt = 0; nt < 2; ++nt) {
        v8f acc = {};
        acc = __builtin_amdgcn_wmma_f32_16x16x32_f16(false, qa[mt][0], false, kb[nt][0], (short)0, acc, false, false);
        acc = __builtin_amdgcn_wmma_f32_16x16x32_f16(false, qa[mt][1], false, kb[nt][1], (short)0, acc, false, false);
#pragma unroll
        for (int r = 0; r < 8; ++r) {
          int i  = mt * 16 + r + 8 * g;
          int jc = nt * 16 + lm;
          float sv = acc[r] + relTab[((i - jc + 31) * 311 + 155) * NH + h];
          if (wH == NWH - 1 && ((i ^ jc) & 16)) sv = -1e9f;  // shifted-window mask
          acc[r] = sv;
        }
        S[mt][nt] = acc;
      }

    // row softmax over 32 keys (row lives in two VGPRs across a 16-lane group)
#pragma unroll
    for (int mt = 0; mt < 2; ++mt)
#pragma unroll
      for (int r = 0; r < 8; ++r) {
        int i = mt * 16 + r + 8 * g;
        float a0 = S[mt][0][r], a1 = S[mt][1][r];
        float mv = fmaxf(a0, a1);
        mv = fmaxf(mv, __shfl_xor(mv, 1));
        mv = fmaxf(mv, __shfl_xor(mv, 2));
        mv = fmaxf(mv, __shfl_xor(mv, 4));
        mv = fmaxf(mv, __shfl_xor(mv, 8));
        float e0 = __expf(a0 - mv), e1 = __expf(a1 - mv);
        float sm = e0 + e1;
        sm += __shfl_xor(sm, 1);
        sm += __shfl_xor(sm, 2);
        sm += __shfl_xor(sm, 4);
        sm += __shfl_xor(sm, 8);
        float inv = 1.f / sm;
        sP[h * 1024 + i * 32 + lm]      = (_Float16)(e0 * inv);
        sP[h * 1024 + i * 32 + 16 + lm] = (_Float16)(e1 * inv);
      }
    __syncthreads();

    // O = P @ V   (M=32, N=64, K=32)
    v16h pa[2];
#pragma unroll
    for (int mt = 0; mt < 2; ++mt)
#pragma unroll
      for (int j = 0; j < 16; ++j)
        pa[mt][j] = sP[h * 1024 + (mt * 16 + lm) * 32 + amapK(j, g)];

#pragma unroll
    for (int nt = 0; nt < 4; ++nt) {
      v16h vb;
#pragma unroll
      for (int j = 0; j < 16; ++j)
        vb[j] = sV[amapK(j, g) * CDIM + h * HD + nt * 16 + lm];
      v8f o0 = {}, o1 = {};
      o0 = __builtin_amdgcn_wmma_f32_16x16x32_f16(false, pa[0], false, vb, (short)0, o0, false, false);
      o1 = __builtin_amdgcn_wmma_f32_16x16x32_f16(false, pa[1], false, vb, (short)0, o1, false, false);
      // scatter into A-swizzled sX (proj-GEMM operand); each head owns its 64 channels
#pragma unroll
      for (int r = 0; r < 8; ++r) {
        int tl = r + 8 * g;                  // token & 15
        int c  = h * HD + nt * 16 + lm;
        int kk = c & 31, ks = c >> 5;
        int ga = (kk >> 3) & 1;
        int j  = (kk & 7) + ((kk & 16) >> 1);
        sX[((ks)      * 32 + tl + 16 * ga) * 16 + j] = (_Float16)o0[r];
        sX[((16 + ks) * 32 + tl + 16 * ga) * 16 + j] = (_Float16)o1[r];
      }
    }
  }
  __syncthreads();

  // ---- Phase 4: Out = O @ Wproj + b, scatter through inverse roll
  for (int t = wave; t < 32; t += 8) {
    v8f acc0 = {}, acc1 = {};
    const _Float16* bb = wProj + (size_t)t * (16 * 32 * 16);
#pragma unroll 4
    for (int kt = 0; kt < 16; ++kt) {
      v16h a0 = *reinterpret_cast<const v16h*>(&sX[((kt) * 32 + lane) * 16]);
      v16h a1 = *reinterpret_cast<const v16h*>(&sX[((16 + kt) * 32 + lane) * 16]);
      v16h b  = *reinterpret_cast<const v16h*>(&bb[(kt * 32 + lane) * 16]);
      acc0 = __builtin_amdgcn_wmma_f32_16x16x32_f16(false, a0, false, b, (short)0, acc0, false, false);
      acc1 = __builtin_amdgcn_wmma_f32_16x16x32_f16(false, a1, false, b, (short)0, acc1, false, false);
    }
    int n = t * 16 + lm;
    float bp = bProj[n];
#pragma unroll
    for (int r = 0; r < 8; ++r) {
      int i0 = r + 8 * g;
      int d0 = (wH * WIN_H + i0 + SHIFT) & (DDIM - 1);
      out[(((size_t)d0) * TDIM + rcol) * CDIM + n] = acc0[r] + bp;
      int i1 = 16 + i0;
      int d1 = (wH * WIN_H + i1 + SHIFT) & (DDIM - 1);
      out[(((size_t)d1) * TDIM + rcol) * CDIM + n] = acc1[r] + bp;
    }
  }
}

extern "C" void kernel_launch(void* const* d_in, const int* in_sizes, int n_in,
                              void* d_out, int out_size, void* d_ws, size_t ws_size,
                              hipStream_t stream) {
  (void)in_sizes; (void)n_in; (void)out_size; (void)ws_size;
  const float* x     = (const float*)d_in[0];
  const float* wqkv  = (const float*)d_in[1];
  const float* wproj = (const float*)d_in[2];
  const float* bproj = (const float*)d_in[3];
  const float* rel   = (const float*)d_in[4];
  const float* qnw   = (const float*)d_in[5];
  const float* knw   = (const float*)d_in[6];
  float* out = (float*)d_out;

  _Float16* wsQKV  = (_Float16*)d_ws;                      // 512*1536 f16 = 1.5 MB
  _Float16* wsProj = wsQKV + (size_t)512 * 1536;           // 512*512  f16 = 0.5 MB

  swizzle_w_f16<<<(512 * 1536) / 256, 256, 0, stream>>>(wqkv, 1536, wsQKV);
  swizzle_w_f16<<<(512 * 512) / 256, 256, 0, stream>>>(wproj, 512, wsProj);
  win_attn_fused<<<dim3(TDIM, NWH), 256, 0, stream>>>(x, wsQKV, wsProj, bproj, rel, qnw, knw, out);
}
